// MultiHeadedAttention_27221502722508
// MI455X (gfx1250) — matrix operand order repaired and checked by reading, not compiled, not run on hardware
//
#include <hip/hip_runtime.h>

typedef __attribute__((ext_vector_type(16))) __bf16 v16bf;
typedef __attribute__((ext_vector_type(8)))  __bf16 bf16x8;
typedef __attribute__((ext_vector_type(4)))  __bf16 bf16x4;
typedef __attribute__((ext_vector_type(8)))  float  v8f;
typedef __attribute__((ext_vector_type(4)))  unsigned int u32x4;
typedef __attribute__((ext_vector_type(8)))  int i32x8;
typedef __attribute__((ext_vector_type(4)))  int i32x4;

#define BATCH   8
#define SEQ     1024
#define DMODEL  1024
#define NHEADS  16
#define HDIM    64

#if defined(__HIP_DEVICE_COMPILE__) && __has_builtin(__builtin_amdgcn_tensor_load_to_lds)
#define HAVE_TDM 1
#else
#define HAVE_TDM 0
#endif

// Combine two contiguous 8-element LDS reads into one WMMA A/B fragment.
__device__ __forceinline__ v16bf mk16(bf16x8 lo, bf16x8 hi) {
    v16bf r;
#pragma unroll
    for (int i = 0; i < 8; ++i) { r[i] = lo[i]; r[i + 8] = hi[i]; }
    return r;
}

#if HAVE_TDM
// Issue a 2D TDM tile load: tileH rows of tileW contiguous bf16 elements,
// row pitch strideElems, packed row-major into LDS at ldsPtr. Tracked by
// TENSORcnt; caller fences with s_wait_tensorcnt + barrier.
__device__ __forceinline__ void tdm_load_2d(const void* gptr, const void* ldsPtr,
                                            unsigned tileW, unsigned tileH,
                                            unsigned strideElems) {
    unsigned long long ga = (unsigned long long)gptr;
    unsigned lds = (unsigned)(unsigned long long)ldsPtr;   // low 32 bits = LDS offset
    u32x4 g0;
    g0[0] = 1u;                                            // count=1, user descriptor
    g0[1] = lds;                                           // lds_addr
    g0[2] = (unsigned)ga;                                  // global_addr[31:0]
    g0[3] = (unsigned)((ga >> 32) & 0x1FFFFFFull) | (2u << 30); // addr[56:32] | type=2
    i32x8 g1;
    g1[0] = 1 << 16;                                       // data_size = 2 bytes
    g1[1] = (int)(tileW << 16);                            // tensor_dim0 (lo16)
    g1[2] = (int)(tileH << 16);                            // tensor_dim1 (lo16)
    g1[3] = (int)(tileW << 16);                            // tile_dim0
    g1[4] = (int)tileH;                                    // tile_dim1 (tile_dim2 = 0)
    g1[5] = (int)strideElems;                              // tensor_dim0_stride (lo32)
    g1[6] = 0;
    g1[7] = 0;
    i32x4 z4 = {0, 0, 0, 0};
#if __clang_major__ >= 23
    i32x8 z8 = {0, 0, 0, 0, 0, 0, 0, 0};
    __builtin_amdgcn_tensor_load_to_lds(g0, g1, z4, z4, z8, 0);
#else
    __builtin_amdgcn_tensor_load_to_lds(g0, g1, z4, z4, 0);
#endif
}
#endif

// ---------------------------------------------------------------------------
// fp32 -> bf16 conversion (grid-stride)
// ---------------------------------------------------------------------------
__global__ __launch_bounds__(256)
void cvt_f32_bf16(const float* __restrict__ in, __bf16* __restrict__ out, int n) {
    int i = blockIdx.x * blockDim.x + threadIdx.x;
    int stride = gridDim.x * blockDim.x;
    for (; i < n; i += stride) out[i] = (__bf16)in[i];
}

// ---------------------------------------------------------------------------
// Weight convert + transpose: fp32 W[K][N] -> bf16 Wt[N][K] (done once; GEMMs
// then stage contiguous rows so all LDS traffic is b128).
// ---------------------------------------------------------------------------
__global__ __launch_bounds__(256)
void wt_cvt_transpose(const float* __restrict__ in, __bf16* __restrict__ out, int K, int N) {
    int i = blockIdx.x * blockDim.x + threadIdx.x;
    int stride = gridDim.x * blockDim.x;
    int total = K * N;
    for (; i < total; i += stride) {
        int k = i / N, n = i - k * N;
        out[(size_t)n * K + k] = (__bf16)in[i];
    }
}

// ---------------------------------------------------------------------------
// V per-head transpose: bf16 Vp[B*S][D] -> Vt[B][H][HD][S]
// ---------------------------------------------------------------------------
__global__ __launch_bounds__(256)
void v_head_transpose(const __bf16* __restrict__ in, __bf16* __restrict__ out) {
    int i = blockIdx.x * blockDim.x + threadIdx.x;
    int stride = gridDim.x * blockDim.x;
    const int total = BATCH * SEQ * DMODEL;
    for (; i < total; i += stride) {
        int tok = i >> 10;            // / DMODEL
        int d   = i & 1023;
        int b   = tok >> 10;          // / SEQ
        int s   = tok & 1023;
        int h   = d >> 6, hd = d & 63;
        out[(((size_t)b * NHEADS + h) * HDIM + hd) * SEQ + s] = in[i];
    }
}

// ---------------------------------------------------------------------------
// C = A[MxK] * Wt[NxK]^T + bias.  Block tile 64x64, 8 waves = 4(M) x 2(N-pair),
// each wave: two 16x16 accumulators sharing the A fragment. K staged 64 at a
// time via TDM (double buffered) or cooperative b128 copies (fallback).
// ---------------------------------------------------------------------------
__global__ __launch_bounds__(256)
void gemm_bf16_wmma(const __bf16* __restrict__ A, const __bf16* __restrict__ Wt,
                    const float* __restrict__ bias,
                    __bf16* __restrict__ Cb, float* __restrict__ Cf,
                    int M, int N, int K) {
    __shared__ __bf16 lA[2][64 * 64];   // [m][k] tiles, 8 KB each
    __shared__ __bf16 lW[2][64 * 64];   // [n][k] tiles (pre-transposed weights)

    const int tid = threadIdx.x, lane = tid & 31, wave = tid >> 5;
    const int l16 = lane & 15, hi = lane >> 4;
    const int mrow = wave >> 1, npair = wave & 1;
    const int tileM = blockIdx.x * 64, tileN = blockIdx.y * 64;
    const int nStages = K >> 6;

    v8f acc0 = {}, acc1 = {};

#if HAVE_TDM
    if (wave == 0) {
        tdm_load_2d(A + (size_t)tileM * K, lA[0], 64, 64, K);
        tdm_load_2d(Wt + (size_t)tileN * K, lW[0], 64, 64, K);
        __builtin_amdgcn_s_wait_tensorcnt(0);
    }
    __syncthreads();
#endif
    for (int s = 0; s < nStages; ++s) {
#if HAVE_TDM
        const int cur = s & 1;
        if (wave == 0 && s + 1 < nStages) {     // prefetch next stage via TDM
            int kc = (s + 1) << 6;
            tdm_load_2d(A + (size_t)tileM * K + kc, lA[cur ^ 1], 64, 64, K);
            tdm_load_2d(Wt + (size_t)tileN * K + kc, lW[cur ^ 1], 64, 64, K);
        }
#else
        const int cur = 0;
        {
            int kc = s << 6;
            int row = tid >> 2, c16 = (tid & 3) * 16;
            *(bf16x8*)(&lA[0][row * 64 + c16])     = *(const bf16x8*)(A + (size_t)(tileM + row) * K + kc + c16);
            *(bf16x8*)(&lA[0][row * 64 + c16 + 8]) = *(const bf16x8*)(A + (size_t)(tileM + row) * K + kc + c16 + 8);
            *(bf16x8*)(&lW[0][row * 64 + c16])     = *(const bf16x8*)(Wt + (size_t)(tileN + row) * K + kc + c16);
            *(bf16x8*)(&lW[0][row * 64 + c16 + 8]) = *(const bf16x8*)(Wt + (size_t)(tileN + row) * K + kc + c16 + 8);
        }
        __syncthreads();
#endif
#pragma unroll
        for (int c = 0; c < 2; ++c) {
            bf16x8 aLo = *(const bf16x8*)(&lA[cur][(mrow * 16 + l16) * 64 + c * 32 + hi * 8]);
            bf16x8 aHi = *(const bf16x8*)(&lA[cur][(mrow * 16 + l16) * 64 + c * 32 + 16 + hi * 8]);
            v16bf av = mk16(aLo, aHi);
            int n0 = npair * 32 + l16;
            bf16x8 b0lo = *(const bf16x8*)(&lW[cur][n0 * 64 + c * 32 + hi * 8]);
            bf16x8 b0hi = *(const bf16x8*)(&lW[cur][n0 * 64 + c * 32 + 16 + hi * 8]);
            acc0 = __builtin_amdgcn_wmma_f32_16x16x32_bf16(
                false, av, false, mk16(b0lo, b0hi), (short)0, acc0, false, false);
            int n1 = n0 + 16;
            bf16x8 b1lo = *(const bf16x8*)(&lW[cur][n1 * 64 + c * 32 + hi * 8]);
            bf16x8 b1hi = *(const bf16x8*)(&lW[cur][n1 * 64 + c * 32 + 16 + hi * 8]);
            acc1 = __builtin_amdgcn_wmma_f32_16x16x32_bf16(
                false, av, false, mk16(b1lo, b1hi), (short)0, acc1, false, false);
        }
#if HAVE_TDM
        if (wave == 0) __builtin_amdgcn_s_wait_tensorcnt(0);
#endif
        __syncthreads();
    }
#pragma unroll
    for (int r = 0; r < 8; ++r) {
        int row = tileM + mrow * 16 + r + hi * 8;
        int col = tileN + npair * 32 + l16;
        float v0 = acc0[r] + bias[col];
        float v1 = acc1[r] + bias[col + 16];
        if (Cf) { Cf[(size_t)row * N + col] = v0; Cf[(size_t)row * N + col + 16] = v1; }
        if (Cb) { Cb[(size_t)row * N + col] = (__bf16)v0; Cb[(size_t)row * N + col + 16] = (__bf16)v1; }
    }
}

// ---------------------------------------------------------------------------
// Fused per-(b, h, 16-query-rows) attention. Q/K tiles and per-head-transposed
// V tiles staged by the TDM; scores kept bf16 in a 16x1024 LDS row buffer so
// softmax is a pure LDS pass; fp32 probs streamed to d_out (the HBM-bound
// part); ctx = P.V with all 8 waves doing WMMA (chunk parity + LDS reduce).
// ---------------------------------------------------------------------------
__global__ __launch_bounds__(256)
void attn_fused(const __bf16* __restrict__ Qp, const __bf16* __restrict__ Kp,
                const __bf16* __restrict__ Vt, const int* __restrict__ mask,
                float* __restrict__ attnP, __bf16* __restrict__ ctx) {
    __shared__ __bf16 sP[16 * 1024];   // 32 KB scores -> probs (bf16)
    __shared__ __bf16 sQ[16 * 64];     // 2 KB
    __shared__ __bf16 sK[128 * 64];    // 16 KB: 8 key tiles per stage
    __shared__ __bf16 sVt[64 * 64];    // 8 KB: V chunk [hd][key]

    const int tid = threadIdx.x, lane = tid & 31, wave = tid >> 5;
    const int l16 = lane & 15, hi = lane >> 4;
    const int q0 = blockIdx.x * 16;
    const int h = blockIdx.y, b = blockIdx.z;

    // ---- stage Q tile 16x64
#if HAVE_TDM
    if (wave == 0) {
        tdm_load_2d(Qp + ((size_t)(b * SEQ + q0)) * DMODEL + h * HDIM, sQ, 64, 16, DMODEL);
        __builtin_amdgcn_s_wait_tensorcnt(0);
    }
#else
    {
        int row = tid >> 4, c4 = (tid & 15) * 4;
        *(bf16x4*)(sQ + row * 64 + c4) =
            *(const bf16x4*)(Qp + ((size_t)(b * SEQ + q0 + row)) * DMODEL + h * HDIM + c4);
    }
#endif
    __syncthreads();

    v16bf aq[2];
#pragma unroll
    for (int c = 0; c < 2; ++c) {
        bf16x8 lo = *(const bf16x8*)(sQ + l16 * 64 + c * 32 + hi * 8);
        bf16x8 hv = *(const bf16x8*)(sQ + l16 * 64 + c * 32 + 16 + hi * 8);
        aq[c] = mk16(lo, hv);
    }

    // ---- scores: stage 128 keys/iter; wave w owns keys [16w, 16w+16)
    for (int kb8 = 0; kb8 < 8; ++kb8) {
#if HAVE_TDM
        if (wave == 0) {
            tdm_load_2d(Kp + ((size_t)(b * SEQ + kb8 * 128)) * DMODEL + h * HDIM, sK, 64, 128, DMODEL);
            __builtin_amdgcn_s_wait_tensorcnt(0);
        }
#else
        {
            int row = tid >> 1, c32 = (tid & 1) * 32;
#pragma unroll
            for (int i = 0; i < 4; ++i)
                *(bf16x8*)(sK + row * 64 + c32 + i * 8) =
                    *(const bf16x8*)(Kp + ((size_t)(b * SEQ + kb8 * 128 + row)) * DMODEL + h * HDIM + c32 + i * 8);
        }
#endif
        __syncthreads();
        v8f sacc = {};
#pragma unroll
        for (int c = 0; c < 2; ++c) {
            int keyrow = wave * 16 + l16;
            bf16x8 blo = *(const bf16x8*)(sK + keyrow * 64 + c * 32 + hi * 8);
            bf16x8 bhi = *(const bf16x8*)(sK + keyrow * 64 + c * 32 + 16 + hi * 8);
            sacc = __builtin_amdgcn_wmma_f32_16x16x32_bf16(
                false, aq[c], false, mk16(blo, bhi), (short)0, sacc, false, false);
        }
#pragma unroll
        for (int r = 0; r < 8; ++r) {
            int qrow = r + hi * 8;
            int key = kb8 * 128 + wave * 16 + l16;
            float s = sacc[r] * 0.125f;                       // 1/sqrt(64)
            int mv = mask[((size_t)b * SEQ + (q0 + qrow)) * SEQ + key];
            if (mv == 0) s = -1e10f;
            sP[qrow * 1024 + key] = (__bf16)s;
        }
        __syncthreads();
    }

    // ---- softmax: wave w handles rows 2w, 2w+1 (wave32 shuffle reductions)
#pragma unroll
    for (int rr = 0; rr < 2; ++rr) {
        int row = wave * 2 + rr;
        float m = -3.0e38f;
        for (int i = 0; i < 32; ++i)
            m = fmaxf(m, (float)sP[row * 1024 + i * 32 + lane]);
#pragma unroll
        for (int off = 16; off > 0; off >>= 1)
            m = fmaxf(m, __shfl_xor(m, off, 32));
        float sum = 0.f;
        for (int i = 0; i < 32; ++i)
            sum += __expf((float)sP[row * 1024 + i * 32 + lane] - m);
#pragma unroll
        for (int off = 16; off > 0; off >>= 1)
            sum += __shfl_xor(sum, off, 32);
        float inv = 1.0f / sum;
        size_t base = (((size_t)b * NHEADS + h) * SEQ + (q0 + row)) * SEQ;
        for (int i = 0; i < 32; ++i) {
            int idx = i * 32 + lane;
            float p = __expf((float)sP[row * 1024 + idx] - m) * inv;
            attnP[base + idx] = p;              // fp32 attention output stream
            sP[row * 1024 + idx] = (__bf16)p;   // bf16 probs for P.V
        }
    }
    __syncthreads();

    // ---- ctx = P(16x1024) . V(1024x64): stage 64 keys/iter of Vt[hd][key];
    // wave w -> hd tile (w&3)*16 for chunk parity (w>>2): all 8 waves do WMMA.
    v8f oacc = {};
    const int hd = (wave & 3) * 16 + l16;
    const int par = wave >> 2;
    const __bf16* vhead = Vt + ((size_t)(b * NHEADS + h) * HDIM) * SEQ;
    for (int ch2 = 0; ch2 < 16; ++ch2) {
#if HAVE_TDM
        if (wave == 0) {
            tdm_load_2d(vhead + ch2 * 64, sVt, 64, 64, SEQ);
            __builtin_amdgcn_s_wait_tensorcnt(0);
        }
#else
        {
            int row = tid >> 2, c16 = (tid & 3) * 16;
            *(bf16x8*)(sVt + row * 64 + c16)     = *(const bf16x8*)(vhead + (size_t)row * SEQ + ch2 * 64 + c16);
            *(bf16x8*)(sVt + row * 64 + c16 + 8) = *(const bf16x8*)(vhead + (size_t)row * SEQ + ch2 * 64 + c16 + 8);
        }
#endif
        __syncthreads();
        int chunk = ch2 * 2 + par;
        int koff = par * 32;
        bf16x8 alo = *(const bf16x8*)(sP + l16 * 1024 + chunk * 32 + hi * 8);
        bf16x8 ahi = *(const bf16x8*)(sP + l16 * 1024 + chunk * 32 + 16 + hi * 8);
        bf16x8 blo = *(const bf16x8*)(sVt + hd * 64 + koff + hi * 8);
        bf16x8 bhi = *(const bf16x8*)(sVt + hd * 64 + koff + 16 + hi * 8);
        oacc = __builtin_amdgcn_wmma_f32_16x16x32_bf16(
            false, mk16(alo, ahi), false, mk16(blo, bhi), (short)0, oacc, false, false);
        __syncthreads();
    }

    // reduce the two parity partials through LDS (reuse sP; safe after barrier)
    float* red = (float*)sP;                   // 16 x 64 fp32
    if (wave >= 4) {
#pragma unroll
        for (int r = 0; r < 8; ++r) {
            int qrow = r + hi * 8;
            red[qrow * 64 + hd] = oacc[r];
        }
    }
    __syncthreads();
    if (wave < 4) {
#pragma unroll
        for (int r = 0; r < 8; ++r) {
            int qrow = r + hi * 8;
            float v = oacc[r] + red[qrow * 64 + hd];
            ctx[((size_t)(b * SEQ + q0 + qrow)) * DMODEL + h * HDIM + hd] = (__bf16)v;
        }
    }
}

// ---------------------------------------------------------------------------
extern "C" void kernel_launch(void* const* d_in, const int* in_sizes, int n_in,
                              void* d_out, int out_size, void* d_ws, size_t ws_size,
                              hipStream_t stream) {
    (void)in_sizes; (void)n_in; (void)out_size; (void)ws_size;
    const float* q    = (const float*)d_in[0];
    const float* k    = (const float*)d_in[1];
    const float* v    = (const float*)d_in[2];
    const int*   mask = (const int*)d_in[3];
    const float* Wq   = (const float*)d_in[4];
    const float* bq   = (const float*)d_in[5];
    const float* Wk   = (const float*)d_in[6];
    const float* bk   = (const float*)d_in[7];
    const float* Wv   = (const float*)d_in[8];
    const float* bv   = (const float*)d_in[9];
    const float* Wo   = (const float*)d_in[10];
    const float* bo   = (const float*)d_in[11];

    const int M = BATCH * SEQ;                     // 8192
    const size_t nTok = (size_t)M * DMODEL;        // 8,388,608
    const size_t nW   = (size_t)DMODEL * DMODEL;   // 1,048,576

    char* w = (char*)d_ws;
    __bf16* qb  = (__bf16*)w; w += nTok * 2;
    __bf16* kb  = (__bf16*)w; w += nTok * 2;
    __bf16* vb  = (__bf16*)w; w += nTok * 2;
    __bf16* Wqt = (__bf16*)w; w += nW * 2;
    __bf16* Wkt = (__bf16*)w; w += nW * 2;
    __bf16* Wvt = (__bf16*)w; w += nW * 2;
    __bf16* Wot = (__bf16*)w; w += nW * 2;
    __bf16* Qp  = (__bf16*)w; w += nTok * 2;
    __bf16* Kp  = (__bf16*)w; w += nTok * 2;
    __bf16* Vp  = (__bf16*)w; w += nTok * 2;
    __bf16* Vth = (__bf16*)w; w += nTok * 2;       // per-head transposed V
    __bf16* ctx = (__bf16*)w; w += nTok * 2;

    float* outP  = (float*)d_out;        // [B,S,D]
    float* attnP = outP + nTok;          // [B,H,S,S]

    // 1) fp32 -> bf16 conversions; weights converted + transposed to [N][K]
    cvt_f32_bf16<<<2048, 256, 0, stream>>>(q, qb, (int)nTok);
    cvt_f32_bf16<<<2048, 256, 0, stream>>>(k, kb, (int)nTok);
    cvt_f32_bf16<<<2048, 256, 0, stream>>>(v, vb, (int)nTok);
    wt_cvt_transpose<<<512, 256, 0, stream>>>(Wq, Wqt, DMODEL, DMODEL);
    wt_cvt_transpose<<<512, 256, 0, stream>>>(Wk, Wkt, DMODEL, DMODEL);
    wt_cvt_transpose<<<512, 256, 0, stream>>>(Wv, Wvt, DMODEL, DMODEL);
    wt_cvt_transpose<<<512, 256, 0, stream>>>(Wo, Wot, DMODEL, DMODEL);

    // 2) QKV projections (TDM-staged WMMA GEMM, bf16 out)
    dim3 gGemm(M / 64, DMODEL / 64);
    gemm_bf16_wmma<<<gGemm, 256, 0, stream>>>(qb, Wqt, bq, Qp, (float*)nullptr, M, DMODEL, DMODEL);
    gemm_bf16_wmma<<<gGemm, 256, 0, stream>>>(kb, Wkt, bk, Kp, (float*)nullptr, M, DMODEL, DMODEL);
    gemm_bf16_wmma<<<gGemm, 256, 0, stream>>>(vb, Wvt, bv, Vp, (float*)nullptr, M, DMODEL, DMODEL);

    // 3) per-head V transpose so P.V B-fragments are contiguous
    v_head_transpose<<<2048, 256, 0, stream>>>(Vp, Vth);

    // 4) fused scores -> softmax -> P.V (writes fp32 attention probs to d_out)
    dim3 gAttn(SEQ / 16, NHEADS, BATCH);
    attn_fused<<<gAttn, 256, 0, stream>>>(Qp, Kp, Vth, mask, attnP, ctx);

    // 5) output projection (fp32 out straight to d_out)
    gemm_bf16_wmma<<<gGemm, 256, 0, stream>>>(ctx, Wot, bo, (__bf16*)nullptr, outP, M, DMODEL, DMODEL);
}
